// SlidingWindowAttention_58119497449719
// MI455X (gfx1250) — matrix operand order repaired and checked by reading, not compiled, not run on hardware
//
#include <hip/hip_runtime.h>
#include <hip/hip_bf16.h>
#include <math.h>

// ---------------------------------------------------------------------------
// Sliding-window attention block for MI455X (gfx1250, wave32, WMMA, TDM).
//   stage 1: qkv = x @ w_qkv^T      (TDM tensor_load_to_lds + WMMA f16 GEMM)
//   stage 2: RoPE on q,k (in-place) (VALU)
//   stage 3: windowed flash attn    (WMMA QK^T + online softmax + WMMA PV)
//   stage 4: y = out @ w_o^T        (TDM + WMMA GEMM)
// Workspace: qkv[4096*6144] f32 + out[4096*2048] f32 = 128 MiB of d_ws.
// ---------------------------------------------------------------------------

typedef __attribute__((ext_vector_type(16))) _Float16 v16h;
typedef __attribute__((ext_vector_type(4)))  _Float16 v4h;
typedef __attribute__((ext_vector_type(8)))  float    v8f;
typedef __attribute__((ext_vector_type(4)))  unsigned u32x4;
typedef __attribute__((ext_vector_type(8)))  int      i32x8;
typedef __attribute__((ext_vector_type(4)))  int      i32x4;

union FragH { v16h v; uint4 u[2]; };

static __device__ __forceinline__ v8f wmma_f16(const FragH& a, const FragH& b, v8f c) {
  return __builtin_amdgcn_wmma_f32_16x16x32_f16(
      /*neg_a=*/false, a.v, /*neg_b=*/false, b.v,
      /*c_mod=*/(short)0, c, /*reuse_a=*/false, /*reuse_b=*/false);
}

// Build a v16h WMMA operand from two 8-float runs in LDS (f32 -> f16).
static __device__ __forceinline__ void frag_cvt(FragH& f, const float* lo, const float* hi) {
  float4 a0 = *(const float4*)(lo);
  float4 a1 = *(const float4*)(lo + 4);
  float4 b0 = *(const float4*)(hi);
  float4 b1 = *(const float4*)(hi + 4);
  v16h v;
  v[0]  = (_Float16)a0.x; v[1]  = (_Float16)a0.y; v[2]  = (_Float16)a0.z; v[3]  = (_Float16)a0.w;
  v[4]  = (_Float16)a1.x; v[5]  = (_Float16)a1.y; v[6]  = (_Float16)a1.z; v[7]  = (_Float16)a1.w;
  v[8]  = (_Float16)b0.x; v[9]  = (_Float16)b0.y; v[10] = (_Float16)b0.z; v[11] = (_Float16)b0.w;
  v[12] = (_Float16)b1.x; v[13] = (_Float16)b1.y; v[14] = (_Float16)b1.z; v[15] = (_Float16)b1.w;
  f.v = v;
}

// ---- Tensor Data Mover: 2-D f32 tile (cols x rows) global -> LDS ----------
// D# per CDNA5 ISA. LDS rows padded by 4 DWORDs via the
// TDM pad feature (pad_interval = 32 DWORDs = one 32-f32 row, pad_amount = 4)
// so LDS row stride is 36 floats (144B, 16B-aligned, bank-conflict friendly).
static __device__ __forceinline__ void tdm_load_tile_f32(
    const float* gsrc, unsigned lds_addr, int rows, int cols, int row_stride_elts) {
  unsigned long long ga = (unsigned long long)(uintptr_t)gsrc;
  u32x4 g0;
  g0[0] = 1u;                                   // count=1 (valid user descriptor)
  g0[1] = lds_addr;                             // LDS byte address
  g0[2] = (unsigned)(ga & 0xFFFFFFFFu);         // global_addr[31:0]
  g0[3] = (unsigned)((ga >> 32) & 0x01FFFFFFu)  // global_addr[56:32]
        | (2u << 30);                           // type = 2 ("image")
  i32x8 g1;
  g1[0] = (int)((2u << 16)                      // data_size = 4 bytes
        | (1u << 20)                            // pad_enable
        | (4u << 22)                            // pad_interval: 32 DWORDs
        | (3u << 25));                          // pad_amount: 4 DWORDs
  g1[1] = (int)((unsigned)cols << 16);          // tensor_dim0[15:0]
  g1[2] = (int)(((unsigned)cols >> 16) | ((unsigned)rows << 16)); // dim0 hi | dim1 lo
  g1[3] = (int)(((unsigned)rows >> 16) | ((unsigned)cols << 16)); // dim1 hi | tile_dim0
  g1[4] = (int)(unsigned)rows;                  // tile_dim1 | tile_dim2(=0)
  g1[5] = (int)(unsigned)row_stride_elts;       // tensor_dim0_stride[31:0]
  g1[6] = 0;                                    // dim0_stride hi | dim1_stride lo
  g1[7] = 0;
  i32x4 z4 = {0, 0, 0, 0};
#if __clang_major__ >= 23
  i32x8 z8 = {0, 0, 0, 0, 0, 0, 0, 0};
  __builtin_amdgcn_tensor_load_to_lds(g0, g1, z4, z4, z8, 0);
#else
  __builtin_amdgcn_tensor_load_to_lds(g0, g1, z4, z4, 0);
#endif
}

// ======================= generic NT GEMM (TDM + WMMA) ======================
// C[M,N] = A[M,K] * B[N,K]^T  (row-major fp32, K innermost on both operands)
// block = 256 threads (8 waves), tile 128x128, K-chunk 32.
#define GT_K 32
#define GT_LDSF 36   // f32 LDS row stride (matches TDM pad config above)

__global__ __launch_bounds__(256)
void swa_gemm_nt(const float* __restrict__ A, const float* __restrict__ Bm,
                 float* __restrict__ C, int M, int N, int K) {
  __shared__ float As32[128 * GT_LDSF];
  __shared__ float Bs32[128 * GT_LDSF];

  const int tid  = threadIdx.x;
  const int lane = tid & 31;
  const int wave = tid >> 5;
  const int half = lane >> 4;
  const int r    = lane & 15;
  const int wm   = wave & 3;   // 4 waves along M
  const int wn   = wave >> 2;  // 2 waves along N
  const int m0   = blockIdx.y * 128;
  const int n0   = blockIdx.x * 128;

  const unsigned lds_a = (unsigned)(uintptr_t)(void*)As32;
  const unsigned lds_b = (unsigned)(uintptr_t)(void*)Bs32;

  v8f zero8 = {0.f,0.f,0.f,0.f,0.f,0.f,0.f,0.f};
  v8f acc[2][4];
#pragma unroll
  for (int tm = 0; tm < 2; ++tm)
#pragma unroll
    for (int tn = 0; tn < 4; ++tn) acc[tm][tn] = zero8;

  for (int kc = 0; kc < K; kc += GT_K) {
    // wave 0 drives the Tensor Data Mover for both tiles, then waits TENSORcnt
    if (tid < 32) {
      tdm_load_tile_f32(A  + (size_t)m0 * K + kc, lds_a, 128, GT_K, K);
      tdm_load_tile_f32(Bm + (size_t)n0 * K + kc, lds_b, 128, GT_K, K);
      __builtin_amdgcn_s_wait_tensorcnt(0);
    }
    __syncthreads();

    // fragments: ds_load_b128 from f32 LDS + v_cvt_pk_f16_f32 into operands
    FragH af[2], bf[4];
#pragma unroll
    for (int tm = 0; tm < 2; ++tm) {
      int row = wm * 32 + tm * 16 + r;
      frag_cvt(af[tm], &As32[row * GT_LDSF + 8 * half],
                       &As32[row * GT_LDSF + 16 + 8 * half]);
    }
#pragma unroll
    for (int tn = 0; tn < 4; ++tn) {
      int row = wn * 64 + tn * 16 + r;
      frag_cvt(bf[tn], &Bs32[row * GT_LDSF + 8 * half],
                       &Bs32[row * GT_LDSF + 16 + 8 * half]);
    }
#pragma unroll
    for (int tm = 0; tm < 2; ++tm)
#pragma unroll
      for (int tn = 0; tn < 4; ++tn)
        acc[tm][tn] = wmma_f16(af[tm], bf[tn], acc[tm][tn]);
    __syncthreads();
  }

  // epilogue: D layout c[i] <-> (M = i + 8*half, N = r)
#pragma unroll
  for (int tm = 0; tm < 2; ++tm)
#pragma unroll
    for (int tn = 0; tn < 4; ++tn)
#pragma unroll
      for (int i = 0; i < 8; ++i) {
        int row = m0 + wm * 32 + tm * 16 + i + 8 * half;
        int col = n0 + wn * 64 + tn * 16 + r;
        C[(size_t)row * N + col] = acc[tm][tn][i];
      }
}

// ======================= RoPE (GPT-NeoX half-split), in place ==============
__global__ __launch_bounds__(256)
void swa_rope(float* __restrict__ qkv) {
  int idx = blockIdx.x * 256 + threadIdx.x;   // [0, 4096*16*64)
  int i = idx & 63;
  int t = idx >> 6;
  int h = t & 15;
  int m = t >> 4;
  if (m >= 4096) return;
  int pos = m & 2047;  // m = b*2048 + n
  float inv = __powf(10000.0f, -(float)i * (1.0f / 64.0f));
  float ang = (float)pos * inv;
  float sn, cs;
  __sincosf(ang, &sn, &cs);
  float* qp = qkv + (size_t)m * 6144 + h * 128;
  float* kp = qp + 2048;
  float q1 = qp[i], q2 = qp[i + 64];
  qp[i]      = q1 * cs - q2 * sn;
  qp[i + 64] = q2 * cs + q1 * sn;
  float k1 = kp[i], k2 = kp[i + 64];
  kp[i]      = k1 * cs - k2 * sn;
  kp[i + 64] = k2 * cs + k1 * sn;
}

// ======================= windowed flash attention ==========================
// one wave (32 threads) per (b, h, 16-query tile); 32-key chunks.
#define ATT_QS 136   // 128 + 8 halves pad (272B rows, 16B-aligned)
#define ATT_KS 136
#define ATT_VS 40    // 32 + 8 halves pad (80B rows)
#define ATT_PS 40

__global__ __launch_bounds__(32)
void swa_attn(const float* __restrict__ qkv, float* __restrict__ out) {
  __shared__ _Float16 Qs[16 * ATT_QS];
  __shared__ _Float16 Ks[32 * ATT_KS];
  __shared__ _Float16 Vt[128 * ATT_VS];  // transposed: [d][j]
  __shared__ _Float16 Ps[16 * ATT_PS];

  const int lane = threadIdx.x;
  const int half = lane >> 4;
  const int r    = lane & 15;

  int bid = blockIdx.x;
  int qt  = bid & 127;        // N/16 = 128 query tiles
  int h   = (bid >> 7) & 15;
  int b   = bid >> 11;
  int q0  = qt * 16;

  const size_t RS = 6144;     // qkv row stride (f32)
  const float* qbase = qkv + (size_t)b * 2048 * RS + (size_t)h * 128;

  // stage Q tile (RoPE already applied): 8 outstanding b128 loads per batch
  {
    const float* src = qbase + (size_t)(q0 + r) * RS + half * 64;
#pragma unroll
    for (int g = 0; g < 2; ++g) {
      float4 t[8];
#pragma unroll
      for (int u = 0; u < 8; ++u) t[u] = *(const float4*)(src + g * 32 + u * 4);
#pragma unroll
      for (int u = 0; u < 8; ++u) {
        v4h hv = {(_Float16)t[u].x, (_Float16)t[u].y, (_Float16)t[u].z, (_Float16)t[u].w};
        *(v4h*)&Qs[r * ATT_QS + half * 64 + g * 32 + u * 4] = hv;
      }
    }
  }
  __syncthreads();

  // hoist Q A-fragments (4 K-chunks of 32 along head dim)
  FragH qa[4];
#pragma unroll
  for (int dk = 0; dk < 4; ++dk) {
    qa[dk].u[0] = *(const uint4*)&Qs[r * ATT_QS + dk * 32 + 8 * half];
    qa[dk].u[1] = *(const uint4*)&Qs[r * ATT_QS + dk * 32 + 16 + 8 * half];
  }

  v8f zero8 = {0.f,0.f,0.f,0.f,0.f,0.f,0.f,0.f};
  v8f acc[8];
#pragma unroll
  for (int tn = 0; tn < 8; ++tn) acc[tn] = zero8;
  float m_i[8], l_i[8];
#pragma unroll
  for (int i = 0; i < 8; ++i) { m_i[i] = -1e30f; l_i[i] = 0.0f; }

  const float scale = 0.08838834764831845f;  // 1/sqrt(128)
  int j_lo = q0 - 511; if (j_lo < 0) j_lo = 0;
  j_lo &= ~31;
  int j_hi = q0 + 15;

  for (int j0 = j_lo; j0 <= j_hi; j0 += 32) {
    // ---- stage K chunk [j][d] and V chunk transposed [d][j] ----
    {
      int j = j0 + lane;  // one key row per lane (always < 2048 here)
      const float* ksrc = qkv + (size_t)(b * 2048 + j) * RS + 2048 + h * 128;
      const float* vsrc = ksrc + 2048;
#pragma unroll
      for (int g = 0; g < 4; ++g) {
        float4 t[8];
#pragma unroll
        for (int u = 0; u < 8; ++u)
          t[u] = *(const float4*)((u < 4 ? ksrc : vsrc) + g * 16 + (u & 3) * 4);
#pragma unroll
        for (int u = 0; u < 4; ++u) {
          v4h hk = {(_Float16)t[u].x, (_Float16)t[u].y, (_Float16)t[u].z, (_Float16)t[u].w};
          *(v4h*)&Ks[lane * ATT_KS + g * 16 + u * 4] = hk;
        }
#pragma unroll
        for (int u = 4; u < 8; ++u) {
          int d = g * 16 + (u & 3) * 4;
          Vt[(d + 0) * ATT_VS + lane] = (_Float16)t[u].x;
          Vt[(d + 1) * ATT_VS + lane] = (_Float16)t[u].y;
          Vt[(d + 2) * ATT_VS + lane] = (_Float16)t[u].z;
          Vt[(d + 3) * ATT_VS + lane] = (_Float16)t[u].w;
        }
      }
      if (j0 + 32 <= j_hi)  // prefetch next key chunk (global_prefetch_b8)
        __builtin_prefetch(ksrc + 32 * RS, 0, 1);
    }
    __syncthreads();

    // ---- scores: two 16x16 tiles, 4 WMMAs each over head dim ----
    v8f s[2];
#pragma unroll
    for (int jt = 0; jt < 2; ++jt) {
      v8f sa = zero8;
      int jl = jt * 16 + r;
#pragma unroll
      for (int dk = 0; dk < 4; ++dk) {
        FragH kb;
        kb.u[0] = *(const uint4*)&Ks[jl * ATT_KS + dk * 32 + 8 * half];
        kb.u[1] = *(const uint4*)&Ks[jl * ATT_KS + dk * 32 + 16 + 8 * half];
        sa = wmma_f16(qa[dk], kb, sa);
      }
      s[jt] = sa;
    }

    // ---- online softmax update (rows at M = i + 8*half, keys at N = r) ----
#pragma unroll
    for (int i = 0; i < 8; ++i) {
      int qi = q0 + i + 8 * half;
      int k0i = j0 + r;
      int k1i = j0 + 16 + r;
      bool v0 = (k0i <= qi) && (k0i + 512 > qi);
      bool v1 = (k1i <= qi) && (k1i + 512 > qi);
      float s0 = v0 ? s[0][i] * scale : -1e30f;
      float s1 = v1 ? s[1][i] * scale : -1e30f;
      float mx = fmaxf(s0, s1);
#pragma unroll
      for (int msk = 1; msk <= 8; msk <<= 1)
        mx = fmaxf(mx, __shfl_xor(mx, msk, 32));
      float m_new = fmaxf(m_i[i], mx);
      float p0 = v0 ? __expf(s0 - m_new) : 0.0f;
      float p1 = v1 ? __expf(s1 - m_new) : 0.0f;
      float sum = p0 + p1;
#pragma unroll
      for (int msk = 1; msk <= 8; msk <<= 1)
        sum += __shfl_xor(sum, msk, 32);
      float corr = __expf(m_i[i] - m_new);
      l_i[i] = l_i[i] * corr + sum;
      m_i[i] = m_new;
      int Mq = i + 8 * half;
      Ps[Mq * ATT_PS + r]      = (_Float16)p0;
      Ps[Mq * ATT_PS + 16 + r] = (_Float16)p1;
#pragma unroll
      for (int tn = 0; tn < 8; ++tn) acc[tn][i] *= corr;
    }
    __syncthreads();

    // ---- P @ V : one K=32 chunk, 8 output tiles along head dim ----
    FragH pf;
    pf.u[0] = *(const uint4*)&Ps[r * ATT_PS + 8 * half];
    pf.u[1] = *(const uint4*)&Ps[r * ATT_PS + 16 + 8 * half];
#pragma unroll
    for (int tn = 0; tn < 8; ++tn) {
      FragH vb;
      int dcol = tn * 16 + r;
      vb.u[0] = *(const uint4*)&Vt[dcol * ATT_VS + 8 * half];
      vb.u[1] = *(const uint4*)&Vt[dcol * ATT_VS + 16 + 8 * half];
      acc[tn] = wmma_f16(pf, vb, acc[tn]);
    }
    __syncthreads();
  }

  // ---- epilogue: out[b, q, h*128 + d] ----
  float* obase = out + (size_t)(b * 2048 + q0) * 2048 + (size_t)h * 128;
#pragma unroll
  for (int tn = 0; tn < 8; ++tn)
#pragma unroll
    for (int i = 0; i < 8; ++i) {
      int Mq = i + 8 * half;
      obase[(size_t)Mq * 2048 + tn * 16 + r] = acc[tn][i] / l_i[i];
    }
}

// ======================= launcher ==========================================
extern "C" void kernel_launch(void* const* d_in, const int* in_sizes, int n_in,
                              void* d_out, int out_size, void* d_ws, size_t ws_size,
                              hipStream_t stream) {
  const float* x     = (const float*)d_in[0];   // [2,2048,2048]
  const float* w_qkv = (const float*)d_in[1];   // [6144,2048]
  const float* w_o   = (const float*)d_in[2];   // [2048,2048]
  float* y = (float*)d_out;                     // [2,2048,2048]

  float* qkv = (float*)d_ws;                    // 4096*6144 f32
  float* att = qkv + (size_t)4096 * 6144;       // 4096*2048 f32

  // 1) qkv = x @ w_qkv^T  (M=4096, N=6144, K=2048)
  swa_gemm_nt<<<dim3(6144 / 128, 4096 / 128), dim3(256), 0, stream>>>(
      x, w_qkv, qkv, 4096, 6144, 2048);

  // 2) RoPE on q,k in place
  swa_rope<<<dim3((4096 * 16 * 64) / 256), dim3(256), 0, stream>>>(qkv);

  // 3) windowed flash attention: one wave per (b,h,16-query tile)
  swa_attn<<<dim3(2 * 16 * 128), dim3(32), 0, stream>>>(qkv, att);

  // 4) y = att @ w_o^T  (M=4096, N=2048, K=2048)
  swa_gemm_nt<<<dim3(2048 / 128, 4096 / 128), dim3(256), 0, stream>>>(
      att, w_o, y, 4096, 2048, 2048);
}
